// MoE_22436909154701
// MI455X (gfx1250) — matrix operand order repaired and checked by reading, not compiled, not run on hardware
//
#include <hip/hip_runtime.h>

// ---------------------------------------------------------------------------
// MoE (dense, all-experts) as two bf16 WMMA GEMMs on gfx1250, with TDM
// (tensor_load_to_lds) double-buffered LDS staging.
//   N = B*S = 16384 tokens, D = 1024, E = 8, H = 128, E*H = 1024.
//   GEMM1: h[N,1024]   = xbf16[N,1024] @ B1[1024,1024]   (B1[k, e*128+h] = W1[e,k,h])
//   epi1 : hs[n,j]     = gate[n, j>>7] * relu(h[n,j] + b1[j])   (bf16)
//   aug  : hs[n,1024+e] = gate[n,e]; hs[n,1032..1055] = 0        (K2 = 1056)
//   GEMM2: out[N,1024] = hs[N,1056] @ B2[1056,1024]
//          B2 rows 0..1023 = W2 flat [E*H, D]; rows 1024..1031 = b2[e,:]; rest 0.
// ---------------------------------------------------------------------------

#define NTOK   16384
#define DMODEL 1024
#define NEXP   8
#define EH     1024      // E*H
#define K2     1056      // EH + 32 (8 gate rows + 24 zero pad)
#define KT1    32        // 1024 / 32
#define KT2    33        // 1056 / 32
#define NTILES 64        // 1024 / 16 output-column tiles (both GEMMs)

typedef __attribute__((ext_vector_type(16))) __bf16        v16bf;
typedef __attribute__((ext_vector_type(8)))  float         v8f;
typedef __attribute__((ext_vector_type(4)))  unsigned int  v4u;
typedef __attribute__((ext_vector_type(8)))  int           v8i_t;
typedef __attribute__((ext_vector_type(4)))  int           v4i_t;

union Frag {                 // one WMMA 16-bit operand fragment per lane: 32 bytes
    v16bf v;
    uint4 q[2];
};

__device__ __forceinline__ unsigned short f32_to_bf16(float f) {
    union { float f; unsigned int u; } c; c.f = f;
    unsigned int u = c.u;
    u += 0x7FFFu + ((u >> 16) & 1u);   // round-to-nearest-even
    return (unsigned short)(u >> 16);
}

// --------------------------------------------------------------- TDM issue
// Build a D# (groups 0/1 per CDNA5 ISA §8) for a 2-D bf16 tile and issue
// TENSOR_LOAD_TO_LDS (6-arg toolchain variant: extra int32x8 group, zeroed
// since our tensors are <= 2-D). Dims/strides in elements.
__device__ __forceinline__ void tdm_load_2d(unsigned int lds_off,
                                            const void*  gptr,
                                            unsigned int td0, unsigned int td1,
                                            unsigned int tile0, unsigned int tile1,
                                            unsigned long long stride0) {
    unsigned long long ga = (unsigned long long)gptr;
    v4u g0;
    g0[0] = 1u;                                             // count=1, user D#
    g0[1] = lds_off;                                        // LDS byte address
    g0[2] = (unsigned int)ga;                               // global addr [31:0]
    g0[3] = (unsigned int)((ga >> 32) & 0x01FFFFFFull)      // global addr [56:32]
          | 0x80000000u;                                    // type = 2 ("image")
    v8i_t g1;
    g1[0] = (int)(1u << 16);                                // data_size = 2 bytes
    g1[1] = (int)((td0 & 0xFFFFu) << 16);                   // tensor_dim0 lo
    g1[2] = (int)(((td0 >> 16) & 0xFFFFu) |                 // tensor_dim0 hi
                  ((td1 & 0xFFFFu) << 16));                 // tensor_dim1 lo
    g1[3] = (int)(((td1 >> 16) & 0xFFFFu) |                 // tensor_dim1 hi
                  ((tile0 & 0xFFFFu) << 16));               // tile_dim0
    g1[4] = (int)(tile1 & 0xFFFFu);                         // tile_dim1 (tile2=0)
    g1[5] = (int)(unsigned int)(stride0 & 0xFFFFFFFFull);   // dim0 stride lo
    g1[6] = (int)(unsigned int)((stride0 >> 32) & 0xFFFFull); // dim0 stride hi
    g1[7] = 0;
    v4i_t z4 = (v4i_t)0;
    v8i_t z8 = (v8i_t)0;
    __builtin_amdgcn_tensor_load_to_lds(g0, g1, z4, z4, z8, 0);
}

// ---------------------------------------------------------------- x -> bf16
__global__ void cvt_x_kernel(const float* __restrict__ x,
                             unsigned short* __restrict__ xb, int n) {
    int i = blockIdx.x * blockDim.x + threadIdx.x;
    if (i < n) xb[i] = f32_to_bf16(x[i]);
}

// ------------------------------------------------- gate softmax + A2 augment
__global__ void gate_kernel(const float* __restrict__ x,
                            const float* __restrict__ Wg,
                            const float* __restrict__ bg,
                            float* __restrict__ gate,
                            unsigned short* __restrict__ hs) {
    int gid = blockIdx.x * blockDim.x + threadIdx.x;   // 0 .. NTOK*8-1
    int tok = gid >> 3;
    int e   = gid & 7;
    if (tok >= NTOK) return;

    const float* xr = x + (size_t)tok * DMODEL;
    float acc = bg[e];
    for (int d = 0; d < DMODEL; ++d)
        acc += xr[d] * Wg[d * NEXP + e];

    float m = acc;
    for (int off = 4; off > 0; off >>= 1)
        m = fmaxf(m, __shfl_xor(m, off, 8));
    float ex = __expf(acc - m);
    float s = ex;
    for (int off = 4; off > 0; off >>= 1)
        s += __shfl_xor(s, off, 8);
    float g = ex / s;

    gate[tok * NEXP + e] = g;
    unsigned short* hrow = hs + (size_t)tok * K2;
    hrow[EH + e] = f32_to_bf16(g);          // gate rows of augmented A2
    int z = EH + NEXP + e * 3;              // 24 zero-pad columns, 3 per thread
    hrow[z + 0] = 0; hrow[z + 1] = 0; hrow[z + 2] = 0;
}

// --------------------------------------------- pack W1 into B-fragment layout
// Packed element (kt, ntile, lane, i) -> B[k = kt*32 + (lane>>4)*16 + i,
//                                         n = ntile*16 + (lane&15)]
__global__ void pack_w1_kernel(const float* __restrict__ W1,
                               unsigned short* __restrict__ pB1) {
    int tid = blockIdx.x * blockDim.x + threadIdx.x;    // KT1*64*32*16 = 1048576
    int i     =  tid        & 15;
    int lane  = (tid >> 4)  & 31;
    int ntile = (tid >> 9)  & 63;
    int kt    =  tid >> 15;
    int n = ntile * 16 + (lane & 15);
    int k = kt * 32 + ((lane >> 4) * 16) + i;
    int e = n >> 7, h = n & 127;
    pB1[tid] = f32_to_bf16(W1[(size_t)e * DMODEL * 128 + (size_t)k * 128 + h]);
}

// ------------------------------------- pack W2 (+ b2 rows + zero pad) for GEMM2
__global__ void pack_w2_kernel(const float* __restrict__ W2,
                               const float* __restrict__ b2,
                               unsigned short* __restrict__ pB2) {
    int tid = blockIdx.x * blockDim.x + threadIdx.x;    // KT2*64*32*16 = 1081344
    if (tid >= KT2 * NTILES * 32 * 16) return;
    int i     =  tid        & 15;
    int lane  = (tid >> 4)  & 31;
    int ntile = (tid >> 9)  & 63;
    int kt    =  tid >> 15;
    int n = ntile * 16 + (lane & 15);
    int k = kt * 32 + ((lane >> 4) * 16) + i;
    float v;
    if (k < EH)            v = W2[(size_t)k * DMODEL + n];          // [E*H, D] flat
    else if (k < EH + 8)   v = b2[(size_t)(k - EH) * DMODEL + n];   // bias rows
    else                   v = 0.0f;
    pB2[tid] = f32_to_bf16(v);
}

// ------------------------------------------------------------- WMMA GEMM core
// 256 threads = 8 waves (2 M x 4 N). WG tile 64x256, wave tile 32x64.
// TDM double-buffers A (64x32) and B (16 KB packed chunk) tiles into LDS.
template<int KT, int LDA, bool RELU_GATE>
__global__ __launch_bounds__(256)
void gemm_kernel(const unsigned short* __restrict__ A,     // [*, LDA] bf16 rows
                 const unsigned short* __restrict__ B,     // packed fragments
                 const float* __restrict__ gate,           // [NTOK, 8]
                 const float* __restrict__ b1,             // [1024]
                 float* __restrict__ outF,                 // GEMM2: [NTOK,1024]
                 unsigned short* __restrict__ outH)        // GEMM1: [NTOK,K2]
{
    __shared__ unsigned short sA[2][64 * 32];      //  4 KB per buffer
    __shared__ unsigned short sB[2][16 * 512];     // 16 KB per buffer

    const int lane = threadIdx.x & 31;
    const int wave = threadIdx.x >> 5;
    const int wm = wave & 1;          // 2 waves along M
    const int wn = wave >> 1;         // 4 waves along N
    const int lrow = lane & 15;
    const int lhi  = lane >> 4;

    const int tokBase = blockIdx.y * 64 + wm * 32;
    const int colBase = blockIdx.x * 256 + wn * 64;

    // TDM global sources for this workgroup
    const unsigned short* gA = A + (size_t)(blockIdx.y * 64) * LDA;      // + kt*32
    const unsigned short* gB = B + (size_t)(blockIdx.x * 16) * 512;      // + kt*32768

    auto issue_stage = [&](int kt, int buf) {
        tdm_load_2d((unsigned int)(unsigned long long)&sA[buf][0],
                    gA + kt * 32,
                    (unsigned int)LDA, (unsigned int)NTOK,   // tensor dims
                    32u, 64u,                                // tile 32 x 64 rows
                    (unsigned long long)LDA);                // row stride
        tdm_load_2d((unsigned int)(unsigned long long)&sB[buf][0],
                    gB + (size_t)kt * (NTILES * 512),
                    0x7FFFFFFFu, 1u,                         // no OOB clipping
                    8192u, 1u,                               // 1-D 16 KB copy
                    8192ull);
    };

    if (threadIdx.x == 0) {
        issue_stage(0, 0);
        if (KT > 1) issue_stage(1, 1);
    }

    v8f acc[2][4];
#pragma unroll
    for (int mi = 0; mi < 2; ++mi)
#pragma unroll
        for (int ni = 0; ni < 4; ++ni)
            acc[mi][ni] = (v8f)(0.0f);

    for (int kt = 0; kt < KT; ++kt) {
        const int buf = kt & 1;

        if (threadIdx.x == 0) {
            // TDM ops from one wave complete in order: <=2 outstanding proves
            // this stage's pair has landed in LDS.
            if (kt + 1 < KT) __builtin_amdgcn_s_wait_tensorcnt(2);
            else             __builtin_amdgcn_s_wait_tensorcnt(0);
        }
        __syncthreads();

        // A fragments from LDS (contiguous K layout matching packed B: lanes 0-15
        // K 0-15, lanes 16-31 K 16-31), rows local to the 64-row tile.
        const unsigned short* ap = &sA[buf][0] + (wm * 32 + lrow) * 32 + lhi * 16;
        Frag a0, a1;
        a0.q[0] = *(const uint4*)(ap);
        a0.q[1] = *(const uint4*)(ap + 8);
        a1.q[0] = *(const uint4*)(ap + 16 * 32);
        a1.q[1] = *(const uint4*)(ap + 16 * 32 + 8);

        // B fragments: 32 contiguous bytes per lane in the packed chunk.
        const unsigned short* bp = &sB[buf][0] + (wn * 4) * 512 + lane * 16;
        Frag bf[4];
#pragma unroll
        for (int ni = 0; ni < 4; ++ni) {
            bf[ni].q[0] = *(const uint4*)(bp + ni * 512);
            bf[ni].q[1] = *(const uint4*)(bp + ni * 512 + 8);
        }

#pragma unroll
        for (int ni = 0; ni < 4; ++ni) {
            acc[0][ni] = __builtin_amdgcn_wmma_f32_16x16x32_bf16(
                false, a0.v, false, bf[ni].v, (short)0, acc[0][ni], false, false);
            acc[1][ni] = __builtin_amdgcn_wmma_f32_16x16x32_bf16(
                false, a1.v, false, bf[ni].v, (short)0, acc[1][ni], false, false);
        }

        __syncthreads();   // everyone done reading buf before TDM overwrites it
        if (threadIdx.x == 0 && kt + 2 < KT)
            issue_stage(kt + 2, buf);
    }

    // Epilogue. C/D layout: element r of v8f -> M = lhi*8 + r, N = lrow.
#pragma unroll
    for (int mi = 0; mi < 2; ++mi) {
#pragma unroll
        for (int ni = 0; ni < 4; ++ni) {
            const int col = colBase + ni * 16 + lrow;
            float bias = 0.0f;
            int e = 0;
            if (RELU_GATE) { bias = b1[col]; e = col >> 7; }
#pragma unroll
            for (int r = 0; r < 8; ++r) {
                const int tok = tokBase + mi * 16 + lhi * 8 + r;
                float v = acc[mi][ni][r];
                if (RELU_GATE) {
                    v += bias;
                    v = v > 0.0f ? v : 0.0f;
                    v *= gate[tok * NEXP + e];
                    outH[(size_t)tok * K2 + col] = f32_to_bf16(v);
                } else {
                    outF[(size_t)tok * DMODEL + col] = v;
                }
            }
        }
    }
}

// ---------------------------------------------------------------------------
extern "C" void kernel_launch(void* const* d_in, const int* in_sizes, int n_in,
                              void* d_out, int out_size, void* d_ws, size_t ws_size,
                              hipStream_t stream) {
    const float* x  = (const float*)d_in[0];
    const float* W1 = (const float*)d_in[1];
    const float* b1 = (const float*)d_in[2];
    const float* W2 = (const float*)d_in[3];
    const float* b2 = (const float*)d_in[4];
    const float* Wg = (const float*)d_in[5];
    const float* bg = (const float*)d_in[6];
    float* out = (float*)d_out;

    char* ws = (char*)d_ws;
    size_t off = 0;
    auto carve = [&](size_t bytes) -> void* {
        void* p = ws + off;
        off = (off + bytes + 255) & ~(size_t)255;
        return p;
    };
    unsigned short* xb   = (unsigned short*)carve((size_t)NTOK * DMODEL * 2);
    unsigned short* hs   = (unsigned short*)carve((size_t)NTOK * K2 * 2);
    float*          gate = (float*)carve((size_t)NTOK * NEXP * 4);
    unsigned short* pB1  = (unsigned short*)carve((size_t)KT1 * NTILES * 32 * 16 * 2);
    unsigned short* pB2  = (unsigned short*)carve((size_t)KT2 * NTILES * 32 * 16 * 2);

    // 1) prep: x->bf16, gate softmax (+A2 augmentation), weight packing
    cvt_x_kernel<<<(NTOK * DMODEL) / 256, 256, 0, stream>>>(x, xb, NTOK * DMODEL);
    gate_kernel<<<(NTOK * NEXP) / 256, 256, 0, stream>>>(x, Wg, bg, gate, hs);
    pack_w1_kernel<<<(KT1 * NTILES * 32 * 16) / 256, 256, 0, stream>>>(W1, pB1);
    pack_w2_kernel<<<(KT2 * NTILES * 32 * 16 + 255) / 256, 256, 0, stream>>>(W2, b2, pB2);

    dim3 gdim(DMODEL / 256, NTOK / 64, 1);   // (4, 256)

    // 2) GEMM1 + fused bias/relu/gate-scale -> hs (bf16)
    gemm_kernel<KT1, DMODEL, true><<<gdim, 256, 0, stream>>>(
        xb, pB1, gate, b1, nullptr, hs);

    // 3) GEMM2 (bias folded into augmented K rows) -> out (f32)
    gemm_kernel<KT2, K2, false><<<gdim, 256, 0, stream>>>(
        hs, pB2, nullptr, nullptr, out, nullptr);
}
